// LongformerAttention_55164559950293
// MI455X (gfx1250) — matrix operand order repaired and checked by reading, not compiled, not run on hardware
//
#include <hip/hip_runtime.h>
#include <hip/hip_bf16.h>

typedef __bf16 bf16;
typedef __attribute__((ext_vector_type(16))) __bf16 v16bf;
typedef __attribute__((ext_vector_type(8)))  __bf16 v8bf;
typedef __attribute__((ext_vector_type(8)))  float  v8f;

#define BD   2
#define SD   2048
#define DD   768
#define HH   12
#define HDD  64
#define WW   256
#define GG   16
#define NEGV (-1.0e9f)

__device__ __forceinline__ v8f wmma_bf16(v16bf a, v16bf b, v8f c) {
  return __builtin_amdgcn_wmma_f32_16x16x32_bf16(false, a, false, b, (short)0, c, false, false);
}

union AF { v16bf v; v8bf h2[2]; };

// ---------------- conversion / transpose / gather ----------------

__global__ __launch_bounds__(256) void cvt_f32_bf16_k(const float* __restrict__ src,
                                                      bf16* __restrict__ dst, int n) {
  int i = blockIdx.x * 256 + threadIdx.x;
  if (i < n) dst[i] = (bf16)src[i];
}

struct WPack { const float* s[7]; bf16* d[7]; };

// Wt[n][k] = W[k][n], f32 -> bf16, 32x32 LDS tiles
__global__ __launch_bounds__(256) void transpose_cvt_k(WPack p) {
  __shared__ float tile[32][33];
  const float* src = p.s[blockIdx.z];
  bf16* dst = p.d[blockIdx.z];
  int bx = blockIdx.x * 32;           // n tile
  int by = blockIdx.y * 32;           // k tile
  int tx = threadIdx.x & 31;
  int ty = threadIdx.x >> 5;          // 0..7
#pragma unroll
  for (int i = 0; i < 4; i++) {
    int k = by + ty + i * 8;
    tile[ty + i * 8][tx] = src[(size_t)k * DD + bx + tx];
  }
  __syncthreads();
#pragma unroll
  for (int i = 0; i < 4; i++) {
    int n = bx + ty + i * 8;
    dst[(size_t)n * DD + by + tx] = (bf16)tile[tx][ty + i * 8];
  }
}

__global__ __launch_bounds__(256) void gather_rows_k(const bf16* __restrict__ xb,
                                                     const int* __restrict__ gidx,
                                                     bf16* __restrict__ xg) {
  int i = blockIdx.x * 256 + threadIdx.x;     // B*G*D = 24576
  if (i >= BD * GG * DD) return;
  int c = i % DD;
  int rr = i / DD;
  int b = rr / GG, g = rr % GG;
  xg[i] = xb[((size_t)b * SD + gidx[g]) * DD + c];
}

// ---------------- pipelined WMMA GEMM (async global->LDS staging) ----------------
// out = (A @ Bt^T + bias)*scale (+resid). A: [M][768] bf16, Bt[n][k] bf16.
// grid (M/128, 768/128), block 256 (8 waves). Each wave: 32M x 64N.
// K-slices (128x32 A, 128x32 B) double-buffered in LDS via
// global_load_async_to_lds_b128, drained with s_wait_asynccnt.

#define KSTEP 32

__global__ __launch_bounds__(256) void gemm_pipe_k(const bf16* __restrict__ A,
                                                   const bf16* __restrict__ Bt,
                                                   const float* __restrict__ bias,
                                                   bf16* __restrict__ outB,
                                                   float* __restrict__ outF,
                                                   const float* __restrict__ resid,
                                                   float scale) {
  __shared__ __attribute__((aligned(64))) bf16 At[2][128][KSTEP];
  __shared__ __attribute__((aligned(64))) bf16 Bs[2][128][KSTEP];
  const int tid = threadIdx.x;
  const int lane = tid & 31, w = tid >> 5;
  const int M0 = blockIdx.x * 128, N0 = blockIdx.y * 128;
  const int mw = w >> 1, nw = w & 1;       // 4 M-waves x 2 N-waves
  const int nCol = lane & 15;
  const int hi8 = (lane >> 4) << 3;        // C-row offset
  const int kb16 = (lane >> 4) << 4;       // B-frag K base

  auto issue = [&](int kc, int buf) {
#pragma unroll
    for (int i = 0; i < 2; i++) {
      int s = tid + i * 256;               // 0..511: 128 rows x 4 x 16B
      int row = s >> 2, part = (s & 3) * 8;
      unsigned lda = (unsigned)(unsigned long long)&At[buf][row][part];
      unsigned long long ga =
          (unsigned long long)(const void*)(A + (size_t)(M0 + row) * DD + kc + part);
      asm volatile("global_load_async_to_lds_b128 %0, %1, off" :: "v"(lda), "v"(ga) : "memory");
      unsigned ldb = (unsigned)(unsigned long long)&Bs[buf][row][part];
      unsigned long long gb =
          (unsigned long long)(const void*)(Bt + (size_t)(N0 + row) * DD + kc + part);
      asm volatile("global_load_async_to_lds_b128 %0, %1, off" :: "v"(ldb), "v"(gb) : "memory");
    }
  };

  v8f acc[2][4] = {};
  issue(0, 0);
  for (int kc = 0; kc < DD; kc += KSTEP) {
    int buf = (kc / KSTEP) & 1;
    if (kc + KSTEP < DD) {
      issue(kc + KSTEP, buf ^ 1);                       // prefetch next slice
      asm volatile("s_wait_asynccnt 0x4" ::: "memory"); // current slice landed
    } else {
      asm volatile("s_wait_asynccnt 0x0" ::: "memory");
    }
    __syncthreads();                                    // all waves' loads visible
    AF a[2];
#pragma unroll
    for (int mt = 0; mt < 2; mt++) {
      const bf16* ap = &At[buf][mw * 32 + mt * 16 + nCol][kb16];
      a[mt].h2[0] = *(const v8bf*)ap;
      a[mt].h2[1] = *(const v8bf*)(ap + 8);
    }
#pragma unroll
    for (int c = 0; c < 4; c++) {
      v16bf bv = *(const v16bf*)&Bs[buf][nw * 64 + c * 16 + nCol][kb16];
#pragma unroll
      for (int mt = 0; mt < 2; mt++)
        acc[mt][c] = wmma_bf16(a[mt].v, bv, acc[mt][c]);
    }
    __syncthreads();                                    // reads done before rewrite
  }
#pragma unroll
  for (int mt = 0; mt < 2; mt++) {
#pragma unroll
    for (int c = 0; c < 4; c++) {
      int col = N0 + nw * 64 + c * 16 + nCol;
      float bvs = bias[col];
#pragma unroll
      for (int r = 0; r < 8; r++) {
        int row = M0 + mw * 32 + mt * 16 + r + hi8;
        float val = (acc[mt][c][r] + bvs) * scale;
        if (resid) val += resid[(size_t)row * DD + col];
        if (outF)  outF[(size_t)row * DD + col] = val;
        if (outB)  outB[(size_t)row * DD + col] = (bf16)val;
      }
    }
  }
}

// ---------------- small direct GEMM (QG: M=32) ----------------
// grid (M/16, 3), block 128 (4 waves, each a 16x64 tile)

__global__ __launch_bounds__(128) void gemm_bf16_k(const bf16* __restrict__ A,
                                                   const bf16* __restrict__ Bt,
                                                   const float* __restrict__ bias,
                                                   bf16* __restrict__ outB,
                                                   float scale) {
  int lane = threadIdx.x & 31;
  int w = threadIdx.x >> 5;
  int M0 = blockIdx.x * 16;
  int N0 = blockIdx.y * 256 + w * 64;
  int nCol = lane & 15;
  int hi8 = (lane >> 4) << 3;
  int kb16 = (lane >> 4) << 4;

  const bf16* arow = A + (size_t)(M0 + nCol) * DD;
  v8f acc[4] = {};
  for (int kc = 0; kc < DD; kc += 32) {
    AF a;
    a.h2[0] = *(const v8bf*)(arow + kc + kb16);
    a.h2[1] = *(const v8bf*)(arow + kc + kb16 + 8);
#pragma unroll
    for (int c = 0; c < 4; c++) {
      const bf16* brow = Bt + (size_t)(N0 + c * 16 + nCol) * DD + kc + kb16;
      v16bf bv = *(const v16bf*)brow;
      acc[c] = wmma_bf16(a.v, bv, acc[c]);
    }
  }
#pragma unroll
  for (int c = 0; c < 4; c++) {
    int col = N0 + c * 16 + nCol;
    float bvs = bias[col];
#pragma unroll
    for (int r = 0; r < 8; r++) {
      int row = M0 + r + hi8;
      float val = (acc[c][r] + bvs) * scale;
      outB[(size_t)row * DD + col] = (bf16)val;
    }
  }
}

// ---------------- local (banded + global-key) attention, one wave per (b,h,qtile) ----------------

__global__ __launch_bounds__(32) void attn_local_k(const bf16* __restrict__ Q,
                                                   const bf16* __restrict__ K,
                                                   const bf16* __restrict__ V,
                                                   const float* __restrict__ mask,
                                                   const int* __restrict__ gidx,
                                                   bf16* __restrict__ attnOut) {
  __shared__ __attribute__((aligned(64))) bf16 Vt[64 * 32];  // [hd][key], keys 16..31 zero pad
  __shared__ __attribute__((aligned(64))) bf16 Pa[16 * 32];  // [m][key],  keys 16..31 zero pad
  const int lane = threadIdx.x;
  const int qt = blockIdx.x, h = blockIdx.y, b = blockIdx.z;
  const int qbase = qt * 16;
  const int nCol = lane & 15;
  const int hi8 = (lane >> 4) << 3;
  const int kb16 = (lane >> 4) << 4;

  for (int i = lane; i < 64 * 16; i += 32) Vt[(i >> 4) * 32 + 16 + (i & 15)] = (bf16)0.0f;
  for (int i = lane; i < 16 * 16; i += 32) Pa[(i >> 4) * 32 + 16 + (i & 15)] = (bf16)0.0f;

  const bf16* qrow = Q + ((size_t)b * SD + qbase + nCol) * DD + h * HDD;
  AF a0, a1;
  a0.h2[0] = *(const v8bf*)(qrow + kb16);
  a0.h2[1] = *(const v8bf*)(qrow + kb16 + 8);
  a1.h2[0] = *(const v8bf*)(qrow + 32 + kb16);
  a1.h2[1] = *(const v8bf*)(qrow + 32 + kb16 + 8);

  float rmax[8], rsum[8];
  v8f o0 = {}, o1 = {}, o2 = {}, o3 = {};
#pragma unroll
  for (int r = 0; r < 8; r++) { rmax[r] = -1e30f; rsum[r] = 0.0f; }
  const float* mrow = mask + (size_t)b * SD;

  auto do_tile = [&](int skA, int skB, bool band, bool negkey) {
    {  // V tile -> LDS transposed: lane covers key=lane>>1, hd half=(lane&1)*32
      const bf16* vp = V + ((size_t)b * SD + skA) * DD + h * HDD + (lane & 1) * 32;
      v16bf v0 = *(const v16bf*)vp;
      v16bf v1 = *(const v16bf*)(vp + 16);
      int key = lane >> 1, hb = (lane & 1) * 32;
#pragma unroll
      for (int j = 0; j < 16; j++) {
        Vt[(hb + j) * 32 + key] = v0[j];
        Vt[(hb + 16 + j) * 32 + key] = v1[j];
      }
    }
    const bf16* kp = K + ((size_t)b * SD + skB) * DD + h * HDD + kb16;
    v16bf bk0 = *(const v16bf*)kp;
    v16bf bk1 = *(const v16bf*)(kp + 32);
    v8f s = {};
    s = wmma_bf16(a0.v, bk0, s);
    s = wmma_bf16(a1.v, bk1, s);
    float mv = mrow[skB];
    float negk = (negkey && mv != 0.0f) ? NEGV : 0.0f;
#pragma unroll
    for (int r = 0; r < 8; r++) {
      int sq = qbase + r + hi8;
      float val = s[r] + negk;
      if (band) { int dd = skB - sq; if (dd < -WW || dd > WW) val = NEGV; }
      s[r] = val;
    }
#pragma unroll
    for (int r = 0; r < 8; r++) {
      float v = s[r];
      float tm = v;
      tm = fmaxf(tm, __shfl_xor(tm, 1, 32));
      tm = fmaxf(tm, __shfl_xor(tm, 2, 32));
      tm = fmaxf(tm, __shfl_xor(tm, 4, 32));
      tm = fmaxf(tm, __shfl_xor(tm, 8, 32));
      float nm = fmaxf(rmax[r], tm);
      float corr = __expf(rmax[r] - nm);
      float p = __expf(v - nm);
      float ts = p;
      ts += __shfl_xor(ts, 1, 32);
      ts += __shfl_xor(ts, 2, 32);
      ts += __shfl_xor(ts, 4, 32);
      ts += __shfl_xor(ts, 8, 32);
      rmax[r] = nm;
      rsum[r] = rsum[r] * corr + ts;
      o0[r] *= corr; o1[r] *= corr; o2[r] *= corr; o3[r] *= corr;
      Pa[(r + hi8) * 32 + nCol] = (bf16)p;
    }
    asm volatile("s_wait_dscnt 0" ::: "memory");
    AF pf;
    pf.h2[0] = *(const v8bf*)(Pa + nCol * 32 + kb16);
    pf.h2[1] = *(const v8bf*)(Pa + nCol * 32 + kb16 + 8);
    v16bf bv0 = *(const v16bf*)(Vt + (0 * 16 + nCol) * 32 + kb16);
    v16bf bv1 = *(const v16bf*)(Vt + (1 * 16 + nCol) * 32 + kb16);
    v16bf bv2 = *(const v16bf*)(Vt + (2 * 16 + nCol) * 32 + kb16);
    v16bf bv3 = *(const v16bf*)(Vt + (3 * 16 + nCol) * 32 + kb16);
    o0 = wmma_bf16(pf.v, bv0, o0);
    o1 = wmma_bf16(pf.v, bv1, o1);
    o2 = wmma_bf16(pf.v, bv2, o2);
    o3 = wmma_bf16(pf.v, bv3, o3);
  };

  do_tile(gidx[lane >> 1], gidx[nCol], false, false);     // s_glob part
  int t0 = qt - 16; if (t0 < 0) t0 = 0;
  int t1 = qt + 16; if (t1 > SD / 16 - 1) t1 = SD / 16 - 1;
  for (int t = t0; t <= t1; ++t)
    do_tile(t * 16 + (lane >> 1), t * 16 + nCol, true, true);

#pragma unroll
  for (int r = 0; r < 8; r++) {
    int sq = qbase + r + hi8;
    float inv = 1.0f / rsum[r];
    if (mrow[sq] < 0.0f) inv = 0.0f;    // p *= ~is_masked
    size_t base = ((size_t)b * SD + sq) * DD + h * HDD;
    attnOut[base + 0 + nCol]  = (bf16)(o0[r] * inv);
    attnOut[base + 16 + nCol] = (bf16)(o1[r] * inv);
    attnOut[base + 32 + nCol] = (bf16)(o2[r] * inv);
    attnOut[base + 48 + nCol] = (bf16)(o3[r] * inv);
  }
}

// ---------------- global-token full attention, one wave per (b,h) ----------------

__global__ __launch_bounds__(32) void attn_globtok_k(const bf16* __restrict__ QG,
                                                     const bf16* __restrict__ KG,
                                                     const bf16* __restrict__ VG,
                                                     const float* __restrict__ mask,
                                                     const int* __restrict__ gidx,
                                                     bf16* __restrict__ attnOut) {
  __shared__ __attribute__((aligned(64))) bf16 Vt[64 * 32];
  __shared__ __attribute__((aligned(64))) bf16 Pa[16 * 32];
  const int lane = threadIdx.x;
  const int h = blockIdx.x, b = blockIdx.y;
  const int nCol = lane & 15;
  const int hi8 = (lane >> 4) << 3;
  const int kb16 = (lane >> 4) << 4;

  for (int i = lane; i < 64 * 16; i += 32) Vt[(i >> 4) * 32 + 16 + (i & 15)] = (bf16)0.0f;
  for (int i = lane; i < 16 * 16; i += 32) Pa[(i >> 4) * 32 + 16 + (i & 15)] = (bf16)0.0f;

  const bf16* qrow = QG + ((size_t)b * GG + nCol) * DD + h * HDD;
  AF a0, a1;
  a0.h2[0] = *(const v8bf*)(qrow + kb16);
  a0.h2[1] = *(const v8bf*)(qrow + kb16 + 8);
  a1.h2[0] = *(const v8bf*)(qrow + 32 + kb16);
  a1.h2[1] = *(const v8bf*)(qrow + 32 + kb16 + 8);

  float rmax[8], rsum[8];
  v8f o0 = {}, o1 = {}, o2 = {}, o3 = {};
#pragma unroll
  for (int r = 0; r < 8; r++) { rmax[r] = -1e30f; rsum[r] = 0.0f; }
  const float* mrow = mask + (size_t)b * SD;

  for (int t = 0; t < SD / 16; ++t) {
    int skA = t * 16 + (lane >> 1);
    int skB = t * 16 + nCol;
    {
      const bf16* vp = VG + ((size_t)b * SD + skA) * DD + h * HDD + (lane & 1) * 32;
      v16bf v0 = *(const v16bf*)vp;
      v16bf v1 = *(const v16bf*)(vp + 16);
      int key = lane >> 1, hb = (lane & 1) * 32;
#pragma unroll
      for (int j = 0; j < 16; j++) {
        Vt[(hb + j) * 32 + key] = v0[j];
        Vt[(hb + 16 + j) * 32 + key] = v1[j];
      }
    }
    const bf16* kp = KG + ((size_t)b * SD + skB) * DD + h * HDD + kb16;
    v16bf bk0 = *(const v16bf*)kp;
    v16bf bk1 = *(const v16bf*)(kp + 32);
    v8f s = {};
    s = wmma_bf16(a0.v, bk0, s);
    s = wmma_bf16(a1.v, bk1, s);
    float negk = (mrow[skB] < 0.0f) ? NEGV : 0.0f;   // is_masked keys
#pragma unroll
    for (int r = 0; r < 8; r++) {
      float v = s[r] + negk;
      float tm = v;
      tm = fmaxf(tm, __shfl_xor(tm, 1, 32));
      tm = fmaxf(tm, __shfl_xor(tm, 2, 32));
      tm = fmaxf(tm, __shfl_xor(tm, 4, 32));
      tm = fmaxf(tm, __shfl_xor(tm, 8, 32));
      float nm = fmaxf(rmax[r], tm);
      float corr = __expf(rmax[r] - nm);
      float p = __expf(v - nm);
      float ts = p;
      ts += __shfl_xor(ts, 1, 32);
      ts += __shfl_xor(ts, 2, 32);
      ts += __shfl_xor(ts, 4, 32);
      ts += __shfl_xor(ts, 8, 32);
      rmax[r] = nm;
      rsum[r] = rsum[r] * corr + ts;
      o0[r] *= corr; o1[r] *= corr; o2[r] *= corr; o3[r] *= corr;
      Pa[(r + hi8) * 32 + nCol] = (bf16)p;
    }
    asm volatile("s_wait_dscnt 0" ::: "memory");
    AF pf;
    pf.h2[0] = *(const v8bf*)(Pa + nCol * 32 + kb16);
    pf.h2[1] = *(const v8bf*)(Pa + nCol * 32 + kb16 + 8);
    v16bf bv0 = *(const v16bf*)(Vt + (0 * 16 + nCol) * 32 + kb16);
    v16bf bv1 = *(const v16bf*)(Vt + (1 * 16 + nCol) * 32 + kb16);
    v16bf bv2 = *(const v16bf*)(Vt + (2 * 16 + nCol) * 32 + kb16);
    v16bf bv3 = *(const v16bf*)(Vt + (3 * 16 + nCol) * 32 + kb16);
    o0 = wmma_bf16(pf.v, bv0, o0);
    o1 = wmma_bf16(pf.v, bv1, o1);
    o2 = wmma_bf16(pf.v, bv2, o2);
    o3 = wmma_bf16(pf.v, bv3, o3);
  }

#pragma unroll
  for (int r = 0; r < 8; r++) {
    int m = r + hi8;
    float inv = 1.0f / rsum[r];
    size_t base = ((size_t)b * SD + gidx[m]) * DD + h * HDD;  // overwrite global rows
    attnOut[base + 0 + nCol]  = (bf16)(o0[r] * inv);
    attnOut[base + 16 + nCol] = (bf16)(o1[r] * inv);
    attnOut[base + 32 + nCol] = (bf16)(o2[r] * inv);
    attnOut[base + 48 + nCol] = (bf16)(o3[r] * inv);
  }
}

// ---------------- LayerNorm ----------------

__global__ __launch_bounds__(256) void layernorm_k(const float* __restrict__ Hb,
                                                   const float* __restrict__ g,
                                                   const float* __restrict__ be,
                                                   float* __restrict__ out) {
  int row = blockIdx.x;
  const float* h = Hb + (size_t)row * DD;
  float s1 = 0.f, s2 = 0.f;
  for (int i = threadIdx.x; i < DD; i += 256) { float v = h[i]; s1 += v; s2 += v * v; }
  __shared__ float r1[256], r2[256];
  r1[threadIdx.x] = s1; r2[threadIdx.x] = s2;
  __syncthreads();
  for (int st = 128; st > 0; st >>= 1) {
    if (threadIdx.x < st) { r1[threadIdx.x] += r1[threadIdx.x + st]; r2[threadIdx.x] += r2[threadIdx.x + st]; }
    __syncthreads();
  }
  float mu = r1[0] * (1.0f / DD);
  float var = r2[0] * (1.0f / DD) - mu * mu;
  float inv = rsqrtf(var + 1e-12f);
  for (int i = threadIdx.x; i < DD; i += 256)
    out[(size_t)row * DD + i] = (h[i] - mu) * inv * g[i] + be[i];
}

// ---------------- launch ----------------

extern "C" void kernel_launch(void* const* d_in, const int* in_sizes, int n_in,
                              void* d_out, int out_size, void* d_ws, size_t ws_size,
                              hipStream_t stream) {
  (void)in_sizes; (void)n_in; (void)out_size; (void)ws_size;
  const float* x    = (const float*)d_in[0];
  const float* mask = (const float*)d_in[1];
  const int*   gidx = (const int*)d_in[2];
  const float* Wq = (const float*)d_in[3];  const float* bq = (const float*)d_in[4];
  const float* Wk = (const float*)d_in[5];  const float* bk = (const float*)d_in[6];
  const float* Wv = (const float*)d_in[7];  const float* bv = (const float*)d_in[8];
  const float* Wqg = (const float*)d_in[9];  const float* bqg = (const float*)d_in[10];
  const float* Wkg = (const float*)d_in[11]; const float* bkg = (const float*)d_in[12];
  const float* Wvg = (const float*)d_in[13]; const float* bvg = (const float*)d_in[14];
  const float* Wo = (const float*)d_in[15];  const float* bo = (const float*)d_in[16];
  const float* gam = (const float*)d_in[17]; const float* bet = (const float*)d_in[18];

  const size_t M = (size_t)BD * SD;           // 4096 rows
  size_t off = 0;
  auto alloc = [&](size_t bytes) {
    void* p = (char*)d_ws + off;
    off += (bytes + 255) & ~(size_t)255;
    return p;
  };
  bf16* xb   = (bf16*)alloc(M * DD * sizeof(bf16));
  bf16* Wt[7];
  for (int i = 0; i < 7; i++) Wt[i] = (bf16*)alloc((size_t)DD * DD * sizeof(bf16));
  bf16* Qb  = (bf16*)alloc(M * DD * sizeof(bf16));
  bf16* Kb  = (bf16*)alloc(M * DD * sizeof(bf16));
  bf16* Vb  = (bf16*)alloc(M * DD * sizeof(bf16));
  bf16* KGb = (bf16*)alloc(M * DD * sizeof(bf16));
  bf16* VGb = (bf16*)alloc(M * DD * sizeof(bf16));
  bf16* xg  = (bf16*)alloc((size_t)BD * GG * DD * sizeof(bf16));
  bf16* QGb = (bf16*)alloc((size_t)BD * GG * DD * sizeof(bf16));
  bf16* attn = (bf16*)alloc(M * DD * sizeof(bf16));
  float* Hb  = (float*)alloc(M * DD * sizeof(float));

  // 1) conversions
  int nx = (int)(M * DD);
  cvt_f32_bf16_k<<<(nx + 255) / 256, 256, 0, stream>>>(x, xb, nx);
  WPack wp;
  wp.s[0] = Wq; wp.s[1] = Wk; wp.s[2] = Wv; wp.s[3] = Wqg; wp.s[4] = Wkg; wp.s[5] = Wvg; wp.s[6] = Wo;
  for (int i = 0; i < 7; i++) wp.d[i] = Wt[i];
  transpose_cvt_k<<<dim3(DD / 32, DD / 32, 7), 256, 0, stream>>>(wp);
  gather_rows_k<<<(BD * GG * DD + 255) / 256, 256, 0, stream>>>(xb, gidx, xg);

  // 2) projections via async-pipelined WMMA GEMM (scale 1/sqrt(HD)=0.125 on q/qg)
  dim3 gp((unsigned)(M / 128), DD / 128);
  gemm_pipe_k<<<gp, 256, 0, stream>>>(xb, Wt[0], bq,  Qb,  nullptr, nullptr, 0.125f);
  gemm_pipe_k<<<gp, 256, 0, stream>>>(xb, Wt[1], bk,  Kb,  nullptr, nullptr, 1.0f);
  gemm_pipe_k<<<gp, 256, 0, stream>>>(xb, Wt[2], bv,  Vb,  nullptr, nullptr, 1.0f);
  gemm_pipe_k<<<gp, 256, 0, stream>>>(xb, Wt[4], bkg, KGb, nullptr, nullptr, 1.0f);
  gemm_pipe_k<<<gp, 256, 0, stream>>>(xb, Wt[5], bvg, VGb, nullptr, nullptr, 1.0f);
  gemm_bf16_k<<<dim3(2, 3), 128, 0, stream>>>(xg, Wt[3], bqg, QGb, 0.125f);

  // 3) banded + global-key attention, then global-token rows overwrite
  attn_local_k<<<dim3(SD / 16, HH, BD), 32, 0, stream>>>(Qb, Kb, Vb, mask, gidx, attn);
  attn_globtok_k<<<dim3(HH, BD), 32, 0, stream>>>(QGb, KGb, VGb, mask, gidx, attn);

  // 4) output projection + bias + residual, then LayerNorm
  gemm_pipe_k<<<gp, 256, 0, stream>>>(attn, Wt[6], bo, nullptr, Hb, x, 1.0f);
  layernorm_k<<<(unsigned)M, 256, 0, stream>>>(Hb, gam, bet, (float*)d_out);
}
